// AGC_86019605004707
// MI455X (gfx1250) — matrix operand order repaired and checked by reading, not compiled, not run on hardware
//
// AGC (AttentiveFP x3 channels) forward for MI455X / gfx1250.
//
// Roofline: ~150 GFLOP of dense GEMM (M up to 65536, K<=512, N<=384) ->
// v_wmma_f32_16x16x32_bf16 (fp32 RNE-rounded to bf16 for A/B, fp32 acc).
// ~3-4 GB of edge gather/scatter traffic -> ~0.15ms at 23.3 TB/s; node
// tables (16MB) live in the 192MB L2, so gathers mostly hit L2.
// wave32: 256-thread blocks = 8 waves; each wave owns TWO 16x16 WMMA tiles
// (shared A fragment), block tile = 32x128, K stepped by 32 through LDS.

#include <hip/hip_runtime.h>
#include <hip/hip_bf16.h>
#include <stdint.h>

// ---------------- static model sizes ----------------
#define NMOLS   2048
#define NATOM   32768
#define EATOM   65536
#define NFRAG   32768
#define EFRAG   65536
#define NFG     16384
#define EJUNC   32768
#define HID     128

typedef __attribute__((ext_vector_type(16))) __bf16 v16bf;
typedef __attribute__((ext_vector_type(8)))  __bf16 v8bf;
typedef __attribute__((ext_vector_type(4)))  __bf16 v4bf;
typedef __attribute__((ext_vector_type(8)))  float  v8f;

// ---------------- device helpers ----------------
__device__ __forceinline__ __bf16 f2bf(float f) {
  unsigned u = __float_as_uint(f);
  unsigned r = u + 0x7fffu + ((u >> 16) & 1u);   // round-to-nearest-even
  unsigned short s = (unsigned short)(r >> 16);
  return __builtin_bit_cast(__bf16, s);
}
__device__ __forceinline__ float actf(float v, int act, float slope) {
  if (act == 1) return v >= 0.f ? v : slope * v;          // leaky relu
  if (act == 2) return v > 0.f ? v : 0.f;                 // relu
  if (act == 3) return v > 0.f ? v : __expf(v) - 1.f;     // elu
  return v;
}
// order-preserving float<->uint map (for atomic max on floats)
__device__ __forceinline__ unsigned ford(float f) {
  unsigned u = __float_as_uint(f);
  return (u & 0x80000000u) ? ~u : (u | 0x80000000u);
}
__device__ __forceinline__ float dford(unsigned u) {
  return (u & 0x80000000u) ? __uint_as_float(u & 0x7fffffffu) : __uint_as_float(~u);
}

// ---------------- WMMA GEMM ----------------
// C[M,N] = ep( (arow? A[arow[m]]: A[m])[K] @ B[K,N] (+Cin) )
// Epilogue: v=acc(+Cin)(+bias); act_pos==1: v=act(v)*scale+shift
//                               act_pos==0: v=act(v*scale+shift)
#define BM 32
#define BN 128
#define BK 32

__global__ __launch_bounds__(256)
void k_gemm(const float* __restrict__ A, int lda, const int* __restrict__ arow,
            const float* __restrict__ B, int ldb,
            float* __restrict__ C, int ldc,
            const float* __restrict__ Cin, int ldcin,
            const float* __restrict__ bias,
            const float* __restrict__ scale,
            const float* __restrict__ shift,
            int M, int N, int K, int act, float slope, int act_pos) {
  __shared__ __align__(16) __bf16 As[BM][BK + 8];   // 32x40 bf16  (2.5 KB)
  __shared__ __align__(16) __bf16 Bs[BN][BK + 8];   // 128x40 bf16 (10 KB), [n][k]
  const int m0 = blockIdx.y * BM;
  const int n0 = blockIdx.x * BN;
  const int tid  = threadIdx.x;
  const int wave = tid >> 5, lane = tid & 31;
  const int wr = wave >> 2, wc = wave & 3;          // 2x4 wave grid
  const int lm = lane & 15, hi = lane >> 4;

  // ---- hoisted staging coordinates (index gather resolved ONCE) ----
  const int sar = tid >> 3;            // A staging row 0..31
  const int sac = (tid & 7) << 2;      // A staging col {0,4,...,28}
  const float* Arow = nullptr;
  {
    int mr = m0 + sar;
    if (mr < M) Arow = A + (size_t)(arow ? arow[mr] : mr) * lda;
  }
  const bool avec = ((lda & 3) == 0);
  const int sbn = (tid & 31) << 2;     // B staging cols {0,4,...,124}
  const bool bvec = ((ldb & 3) == 0);

  v8f acc0 = {}, acc1 = {};

  for (int k0 = 0; k0 < K; k0 += BK) {
    // ---- stage A: one float4 per thread, pack to 4 bf16 (8B LDS store) ----
    {
      float a0 = 0.f, a1 = 0.f, a2 = 0.f, a3 = 0.f;
      if (Arow) {
        int kc = k0 + sac;
        if (avec && (kc + 4 <= K)) {
          float4 t = *(const float4*)(Arow + kc);
          a0 = t.x; a1 = t.y; a2 = t.z; a3 = t.w;
        } else {
          if (kc + 0 < K) a0 = Arow[kc + 0];
          if (kc + 1 < K) a1 = Arow[kc + 1];
          if (kc + 2 < K) a2 = Arow[kc + 2];
          if (kc + 3 < K) a3 = Arow[kc + 3];
        }
      }
      v4bf pk = { f2bf(a0), f2bf(a1), f2bf(a2), f2bf(a3) };
      *(v4bf*)&As[sar][sac] = pk;
    }
    // ---- stage B transposed: four float4 loads per thread ----
    #pragma unroll
    for (int j = 0; j < 4; ++j) {
      int kk = (tid >> 5) + j * 8;     // 0..31
      float b0 = 0.f, b1 = 0.f, b2 = 0.f, b3 = 0.f;
      if (k0 + kk < K) {
        const float* Brow = B + (size_t)(k0 + kk) * ldb + n0 + sbn;
        if (bvec && (n0 + sbn + 4 <= N)) {
          float4 t = *(const float4*)Brow;
          b0 = t.x; b1 = t.y; b2 = t.z; b3 = t.w;
        } else {
          if (n0 + sbn + 0 < N) b0 = Brow[0];
          if (n0 + sbn + 1 < N) b1 = Brow[1];
          if (n0 + sbn + 2 < N) b2 = Brow[2];
          if (n0 + sbn + 3 < N) b3 = Brow[3];
        }
      }
      Bs[sbn + 0][kk] = f2bf(b0);
      Bs[sbn + 1][kk] = f2bf(b1);
      Bs[sbn + 2][kk] = f2bf(b2);
      Bs[sbn + 3][kk] = f2bf(b3);
    }
    // prefetch next K-chunk (global_prefetch_b8)
    if (k0 + BK < K) {
      if (Arow && (k0 + BK + sac) < K)
        __builtin_prefetch(Arow + k0 + BK + sac, 0, 1);
      int kk = tid >> 5;
      if ((k0 + BK + kk) < K && (n0 + sbn) < N)
        __builtin_prefetch(&B[(size_t)(k0 + BK + kk) * ldb + n0 + sbn], 0, 1);
    }
    __syncthreads();
    // ---- fragments in contiguous k order (same for A and B) ----
    // A(16x32): lane<16 holds K 0..15      ; lane>=16 holds K 16..31
    // B(32x16): lane<16 holds K 0..15      ; lane>=16 holds K 16..31
    union { v16bf v; v8bf h[2]; } af, b0f, b1f;
    const __bf16* ap = &As[wr * 16 + lm][0];
    const int ka = hi ? 16 : 0;
    af.h[0] = *(const v8bf*)(ap + ka);
    af.h[1] = *(const v8bf*)(ap + ka + 8);
    const int kb = hi ? 16 : 0;
    const __bf16* bp0 = &Bs[wc * 32 + lm][0];
    b0f.h[0] = *(const v8bf*)(bp0 + kb);
    b0f.h[1] = *(const v8bf*)(bp0 + kb + 8);
    const __bf16* bp1 = &Bs[wc * 32 + 16 + lm][0];
    b1f.h[0] = *(const v8bf*)(bp1 + kb);
    b1f.h[1] = *(const v8bf*)(bp1 + kb + 8);
    acc0 = __builtin_amdgcn_wmma_f32_16x16x32_bf16(false, af.v, false, b0f.v,
                                                   (short)0, acc0, false, false);
    acc1 = __builtin_amdgcn_wmma_f32_16x16x32_bf16(false, af.v, false, b1f.v,
                                                   (short)0, acc1, false, false);
    __syncthreads();
  }
  // ---- epilogue; C layout: lane n = lm, acc[r] -> row wr*16 + hi*8 + r ----
  #pragma unroll
  for (int t = 0; t < 2; ++t) {
    v8f acc = t ? acc1 : acc0;
    const int n = n0 + wc * 32 + t * 16 + lm;
    if (n < N) {
      const float bs = bias ? bias[n] : 0.f;
      const float sc = scale ? scale[n] : 1.f;
      const float sh = shift ? shift[n] : 0.f;
      #pragma unroll
      for (int r = 0; r < 8; ++r) {
        int m = m0 + wr * 16 + hi * 8 + r;
        if (m < M) {
          float v = acc[r];
          if (Cin) v += Cin[(size_t)m * ldcin + n];
          v += bs;
          if (act_pos == 1) v = actf(v, act, slope);
          v = v * sc + sh;
          if (act_pos == 0) v = actf(v, act, slope);
          C[(size_t)m * ldc + n] = v;
        }
      }
    }
  }
}

// ---------------- elementwise / segment kernels (feature dim fixed = 128) ----
__global__ void k_fill_f32(float* p, float v, int n) {
  int i = blockIdx.x * blockDim.x + threadIdx.x;
  if (i < n) p[i] = v;
}
__global__ void k_fill_u32(unsigned* p, unsigned v, int n) {
  int i = blockIdx.x * blockDim.x + threadIdx.x;
  if (i < n) p[i] = v;
}
// alpha = g*rsqrt(1+eps); shift = (b? b*alpha:0) + beta
__global__ void k_prep_bn(const float* b, const float* g, const float* beta,
                          float* alpha, float* shift, int n) {
  int i = blockIdx.x * blockDim.x + threadIdx.x;
  if (i < n) {
    float a = g[i] * rsqrtf(1.f + 1e-5f);
    alpha[i] = a;
    shift[i] = (b ? b[i] * a : 0.f) + beta[i];
  }
}
__global__ __launch_bounds__(256)
void k_rowdot(const float* __restrict__ x, const float* __restrict__ w,
              float* __restrict__ y, int rows) {     // cols == ld == 128
  int row = blockIdx.x * 8 + (threadIdx.x >> 5);
  int lane = threadIdx.x & 31;
  if (row >= rows) return;
  const float* xr = x + (size_t)row * HID;
  float s = xr[lane] * w[lane] + xr[32 + lane] * w[32 + lane] +
            xr[64 + lane] * w[64 + lane] + xr[96 + lane] * w[96 + lane];
  for (int off = 16; off; off >>= 1) s += __shfl_xor(s, off, 32);
  if (lane == 0) y[row] = s;
}
__global__ void k_edge_score(const float* s1, const int* i1, const float* s2,
                             const int* i2, float* out, int n, float slope) {
  int e = blockIdx.x * blockDim.x + threadIdx.x;
  if (e >= n) return;
  float a = s1[i1 ? i1[e] : e] + s2[i2 ? i2[e] : e];
  out[e] = a >= 0.f ? a : slope * a;
}
__global__ void k_segmax(const float* a, const int* seg, unsigned* mx, int n) {
  int i = blockIdx.x * blockDim.x + threadIdx.x;
  if (i < n) atomicMax(&mx[seg[i]], ford(a[i]));
}
__global__ void k_exps(const float* a, const int* seg, const unsigned* mx,
                       float* e, float* sum, int n) {
  int i = blockIdx.x * blockDim.x + threadIdx.x;
  if (i >= n) return;
  float m = dford(mx[seg[i]]);
  if (!(m > -1e37f && m < 1e37f)) m = 0.f;   // isfinite fix (empty segs)
  float ev = __expf(a[i] - m);
  e[i] = ev;
  atomicAdd(&sum[seg[i]], ev);
}
__global__ void k_segnorm(const float* e, const int* seg, const float* sum,
                          float* out, int n) {
  int i = blockIdx.x * blockDim.x + threadIdx.x;
  if (i < n) out[i] = e[i] / (sum[seg[i]] + 1e-16f);
}
__global__ void k_bcast_row(float* out, int ldo, const float* vec, int rows) {
  int i = blockIdx.x * blockDim.x + threadIdx.x;
  if (i >= rows * HID) return;
  out[(size_t)(i >> 7) * ldo + (i & 127)] = vec[i & 127];
}
// out[seg[it]][c] += (w? w[it]:1) * x[(gidx? gidx[it]:it)][c]   (c in 0..127)
__global__ void k_scatter(const float* __restrict__ x, int ldx, const int* gidx,
                          const float* w, const int* __restrict__ seg,
                          float* out, int ldo, int items) {
  int i = blockIdx.x * blockDim.x + threadIdx.x;
  if (i >= items * HID) return;
  int it = i >> 7, c = i & 127;
  int xr = gidx ? gidx[it] : it;
  float v = x[(size_t)xr * ldx + c];
  if (w) v *= w[it];
  atomicAdd(&out[(size_t)seg[it] * ldo + c], v);
}
__global__ void k_act(const float* x, float* y, int n, int act, float slope) {
  int i = blockIdx.x * blockDim.x + threadIdx.x;
  if (i < n) y[i] = actf(x[i], act, slope);
}
__global__ void k_gru(const float* __restrict__ gi, const float* __restrict__ gh,
                      float* __restrict__ h, int rows) {
  int i = blockIdx.x * blockDim.x + threadIdx.x;
  if (i >= rows * HID) return;
  int r = i >> 7, c = i & 127;
  const float* pi = gi + (size_t)r * 3 * HID;
  const float* ph = gh + (size_t)r * 3 * HID;
  float rr = 1.f / (1.f + __expf(-(pi[c] + ph[c])));
  float zz = 1.f / (1.f + __expf(-(pi[HID + c] + ph[HID + c])));
  float nn = tanhf(pi[2 * HID + c] + rr * ph[2 * HID + c]);
  float hv = h[(size_t)r * HID + c];
  h[(size_t)r * HID + c] = (1.f - zz) * nn + zz * hv;
}
__global__ void k_mean_relu4(const float* src, size_t hstride, float* dst,
                             int ldd, int rows) {
  int i = blockIdx.x * blockDim.x + threadIdx.x;
  if (i >= rows * HID) return;
  int r = i >> 7, c = i & 127;
  float s = 0.f;
  for (int h = 0; h < 4; ++h) s += src[h * hstride + (size_t)r * HID + c];
  s *= 0.25f;
  dst[(size_t)r * ldd + c] = s > 0.f ? s : 0.f;
}
__global__ void k_copy(const float* s, int lds_, float* d, int ldd, int rows) {
  int i = blockIdx.x * blockDim.x + threadIdx.x;
  if (i >= rows * HID) return;
  int r = i >> 7, c = i & 127;
  d[(size_t)r * ldd + c] = s[(size_t)r * lds_ + c];
}

// ---------------- host orchestration ----------------
struct AfpP {
  const float *lin_w, *lin_b, *g1_w, *g1_b, *g2_w, *g2_b, *g_att_l, *g_att_r,
      *g_bias, *conv_w, *att_src, *att_dst, *conv_b, *gru_wih, *gru_whh,
      *gru_bih, *gru_bhh, *mol_w, *mol_att_src, *mol_att_dst, *mol_bias,
      *mgru_wih, *mgru_whh, *mgru_bih, *mgru_bhh, *out_w, *out_b;
};
static inline int cdiv(int a, int b) { return (a + b - 1) / b; }

extern "C" void kernel_launch(void* const* d_in, const int* in_sizes, int n_in,
                              void* d_out, int out_size, void* d_ws, size_t ws_size,
                              hipStream_t stream) {
  (void)in_sizes; (void)n_in; (void)out_size; (void)ws_size;
  auto Pf = [&](int i) { return (const float*)d_in[i]; };
  auto Pi = [&](int i) { return (const int*)d_in[i]; };

  // ---- inputs ----
  const float* atom_x   = Pf(0);
  const int*   asrc     = Pi(1);
  const int*   adst     = Pi(1) + EATOM;
  const float* atom_ea  = Pf(2);
  const int*   abatch   = Pi(3);
  const float* frag_x   = Pf(4);
  const int*   fsrc     = Pi(5);
  const int*   fdst     = Pi(5) + EFRAG;
  const float* frag_ea  = Pf(6);
  const int*   fbatch   = Pi(7);
  const float* junc_x   = Pf(8);
  const int*   jsrc     = Pi(9);
  const int*   jdst     = Pi(9) + EJUNC;
  const float* junc_ea  = Pf(10);
  const int*   jbatch   = Pi(11);

  // ---- params: JAX pytree flattening (nested dicts, sorted keys) ----
  // top: frag, head, junc, origin
  // frag.att{b,beta,g,w}=12..15, frag.heads(27)=16..42
  // head{b1,beta1,bm,bo,g1,w1,wm,wo}=43..50
  // junc.edge=51..54, junc.heads=55..81, junc.node=82..85, proj_b=86, proj_w=87
  // origin.att=88..91, origin.edge=92..95, origin.heads=96..122, origin.node=123..126
  auto make_head = [&](int base, int h, int nin, int ein) -> AfpP {
    const int ne = 2, nl = 3;
    AfpP p;
    p.att_dst  = Pf(base + 0)  + (size_t)h * ne * HID;
    p.att_src  = Pf(base + 1)  + (size_t)h * ne * HID;
    p.conv_b   = Pf(base + 2)  + (size_t)h * ne * HID;
    p.conv_w   = Pf(base + 3)  + (size_t)h * ne * HID * HID;
    p.g1_b     = Pf(base + 4)  + (size_t)h * HID;
    p.g1_w     = Pf(base + 5)  + (size_t)h * (HID + ein) * HID;
    p.g2_b     = Pf(base + 6)  + (size_t)h * HID;
    p.g2_w     = Pf(base + 7)  + (size_t)h * HID * HID;
    p.g_att_l  = Pf(base + 8)  + (size_t)h * HID;
    p.g_att_r  = Pf(base + 9)  + (size_t)h * HID;
    p.g_bias   = Pf(base + 10) + (size_t)h * HID;
    p.gru_bhh  = Pf(base + 11) + (size_t)h * nl * 3 * HID;
    p.gru_bih  = Pf(base + 12) + (size_t)h * nl * 3 * HID;
    p.gru_whh  = Pf(base + 13) + (size_t)h * nl * HID * 3 * HID;
    p.gru_wih  = Pf(base + 14) + (size_t)h * nl * HID * 3 * HID;
    p.lin_b    = Pf(base + 15) + (size_t)h * HID;
    p.lin_w    = Pf(base + 16) + (size_t)h * nin * HID;
    p.mgru_bhh = Pf(base + 17) + (size_t)h * 3 * HID;
    p.mgru_bih = Pf(base + 18) + (size_t)h * 3 * HID;
    p.mgru_whh = Pf(base + 19) + (size_t)h * HID * 3 * HID;
    p.mgru_wih = Pf(base + 20) + (size_t)h * HID * 3 * HID;
    p.mol_att_dst = Pf(base + 21) + (size_t)h * HID;
    p.mol_att_src = Pf(base + 22) + (size_t)h * HID;
    p.mol_bias    = Pf(base + 23) + (size_t)h * HID;
    p.mol_w       = Pf(base + 24) + (size_t)h * HID * HID;
    p.out_b       = Pf(base + 25) + (size_t)h * HID;
    p.out_w       = Pf(base + 26) + (size_t)h * HID * HID;
    return p;
  };

  // ---- workspace carve-up (assumes generous ws; ~490MB total) ----
  size_t off = 0;
  float* ws = (float*)d_ws;
  auto alloc = [&](size_t n) { float* p = ws + off; off += n; return p; };
  float* NX   = alloc((size_t)NATOM * HID);
  float* NEA  = alloc((size_t)EATOM * HID);
  float* H    = alloc((size_t)NATOM * HID);
  float* Xb   = alloc((size_t)NATOM * HID);
  float* AGG  = alloc((size_t)NATOM * HID);
  float* Gb   = alloc((size_t)NATOM * HID);
  float* GIb  = alloc((size_t)NATOM * 3 * HID);
  float* GHb  = alloc((size_t)NATOM * 3 * HID);
  float* Mb   = alloc((size_t)EATOM * HID);
  float* T2   = alloc((size_t)EATOM * HID);
  float* Ab   = alloc(EATOM);
  float* Eb   = alloc(EATOM);
  float* S1   = alloc(EATOM);
  float* S2   = alloc(NATOM);
  float* SMX  = alloc(NATOM);     // used as unsigned
  float* SSM  = alloc(NATOM);
  float* GOUT = alloc((size_t)NFG * HID);
  float* GD   = alloc((size_t)NFG * HID);
  float* GAGG = alloc((size_t)NFG * HID);
  float* GX   = alloc((size_t)NFG * HID);
  float* GGI  = alloc((size_t)NFG * 3 * HID);
  float* GGH  = alloc((size_t)NFG * 3 * HID);
  float* CAT  = alloc((size_t)NFG * 4 * HID);
  float* GFRAG= alloc((size_t)NFG * HID);
  float* JX   = alloc((size_t)NFG * 2 * HID);
  float* JEA  = alloc((size_t)EJUNC * HID);
  float* JH   = alloc((size_t)NFG * HID);
  float* JLAT = alloc((size_t)4 * NMOLS * HID);
  float* FEAT = alloc((size_t)NMOLS * 384);
  float* Dh   = alloc((size_t)NMOLS * 192);
  float* H2   = alloc((size_t)NMOLS * 192);
  float* EP   = alloc(16 * 256);  // epilogue scale/shift slots
  int ep = 0;
  auto epslot = [&]() { float* p = EP + (ep++) * 256; return p; };

  // ---- launch helpers ----
  auto gemm = [&](const float* A, int lda, const int* arow, const float* B, int ldb,
                  float* C, int ldc, const float* Cin, int ldcin,
                  const float* bias, const float* scale, const float* shift,
                  int M, int N, int K, int act, float slope, int act_pos) {
    dim3 g(cdiv(N, BN), cdiv(M, BM));
    k_gemm<<<g, 256, 0, stream>>>(A, lda, arow, B, ldb, C, ldc, Cin, ldcin,
                                  bias, scale, shift, M, N, K, act, slope, act_pos);
  };
  auto fillf = [&](float* p, float v, int n) {
    k_fill_f32<<<cdiv(n, 256), 256, 0, stream>>>(p, v, n);
  };
  auto rowdot = [&](const float* x, const float* w, float* y, int rows) {
    k_rowdot<<<cdiv(rows, 8), 256, 0, stream>>>(x, w, y, rows);
  };
  auto escore = [&](const float* s1, const int* i1, const float* s2, const int* i2,
                    float* o, int n, float sl) {
    k_edge_score<<<cdiv(n, 256), 256, 0, stream>>>(s1, i1, s2, i2, o, n, sl);
  };
  auto segsoftmax = [&](float* a, const int* seg, int items, int nseg) {
    k_fill_u32<<<cdiv(nseg, 256), 256, 0, stream>>>((unsigned*)SMX, 0x007FFFFFu, nseg); // ord(-inf)
    fillf(SSM, 0.f, nseg);
    k_segmax<<<cdiv(items, 256), 256, 0, stream>>>(a, seg, (unsigned*)SMX, items);
    k_exps<<<cdiv(items, 256), 256, 0, stream>>>(a, seg, (unsigned*)SMX, Eb, SSM, items);
    k_segnorm<<<cdiv(items, 256), 256, 0, stream>>>(Eb, seg, SSM, a, items);
  };
  auto bcast = [&](float* o, int ldo, const float* v, int rows) {
    k_bcast_row<<<cdiv(rows * HID, 256), 256, 0, stream>>>(o, ldo, v, rows);
  };
  auto scatter = [&](const float* x, int ldx, const int* g, const float* w,
                     const int* seg, float* o, int ldo, int items) {
    k_scatter<<<cdiv(items * HID, 256), 256, 0, stream>>>(x, ldx, g, w, seg, o, ldo, items);
  };
  auto actk = [&](const float* x, float* y, int n, int a, float sl) {
    k_act<<<cdiv(n, 256), 256, 0, stream>>>(x, y, n, a, sl);
  };
  auto gruk = [&](const float* gi, const float* gh, float* h, int rows) {
    k_gru<<<cdiv(rows * HID, 256), 256, 0, stream>>>(gi, gh, h, rows);
  };
  auto bnprep = [&](const float* b, const float* g, const float* beta, int n,
                    const float** sc, const float** sh) {
    float* a = epslot(); float* s = epslot();
    k_prep_bn<<<cdiv(n, 256), 256, 0, stream>>>(b, g, beta, a, s, n);
    *sc = a; *sh = s;
  };

  // ---- one AttentiveFP head ----
  auto afp_run = [&](const AfpP& p, const float* x, int nin, const int* src,
                     const int* dst, const float* eattr, int ein, const int* batch,
                     int Nn, int Ee, int Gg, float* outp, int out_ld) {
    // h = lrelu(x @ lin_w + lin_b)
    gemm(x, nin, nullptr, p.lin_w, HID, H, HID, nullptr, 0, p.lin_b, nullptr, nullptr,
         Nn, HID, nin, 1, 0.01f, 0);
    // GATEConv: m = lrelu([h[src], eattr] @ g1_w + g1_b)  (split-K, gather-GEMM)
    gemm(H, HID, src, p.g1_w, HID, Mb, HID, nullptr, 0, nullptr, nullptr, nullptr,
         Ee, HID, HID, 0, 0.f, 0);
    gemm(eattr, ein, nullptr, p.g1_w + HID * HID, HID, Mb, HID, Mb, HID,
         p.g1_b, nullptr, nullptr, Ee, HID, ein, 1, 0.01f, 0);
    rowdot(Mb, p.g_att_l, S1, Ee);
    rowdot(H, p.g_att_r, S2, Nn);
    escore(S1, nullptr, S2, dst, Ab, Ee, 0.01f);
    segsoftmax(Ab, dst, Ee, Nn);
    gemm(Mb, HID, nullptr, p.g2_w, HID, T2, HID, nullptr, 0, p.g2_b, nullptr, nullptr,
         Ee, HID, HID, 0, 0.f, 0);
    bcast(AGG, HID, p.g_bias, Nn);
    scatter(T2, HID, nullptr, Ab, dst, AGG, HID, Ee);
    actk(AGG, Xb, Nn * HID, 3, 0.f);
    gemm(Xb, HID, nullptr, p.gru_wih, 3 * HID, GIb, 3 * HID, nullptr, 0,
         p.gru_bih, nullptr, nullptr, Nn, 3 * HID, HID, 0, 0.f, 0);
    gemm(H, HID, nullptr, p.gru_whh, 3 * HID, GHb, 3 * HID, nullptr, 0,
         p.gru_bhh, nullptr, nullptr, Nn, 3 * HID, HID, 0, 0.f, 0);
    gruk(GIb, GHb, H, Nn);
    // GATConv layers
    for (int l = 0; l < 2; ++l) {
      gemm(H, HID, nullptr, p.conv_w + (size_t)l * HID * HID, HID, Gb, HID,
           nullptr, 0, nullptr, nullptr, nullptr, Nn, HID, HID, 0, 0.f, 0);
      rowdot(Gb, p.att_src + l * HID, S1, Nn);
      rowdot(Gb, p.att_dst + l * HID, S2, Nn);
      escore(S1, src, S2, dst, Ab, Ee, 0.01f);
      segsoftmax(Ab, dst, Ee, Nn);
      bcast(AGG, HID, p.conv_b + l * HID, Nn);
      scatter(Gb, HID, src, Ab, dst, AGG, HID, Ee);
      actk(AGG, Xb, Nn * HID, 3, 0.f);
      gemm(Xb, HID, nullptr, p.gru_wih + (size_t)(l + 1) * HID * 3 * HID, 3 * HID,
           GIb, 3 * HID, nullptr, 0, p.gru_bih + (l + 1) * 3 * HID, nullptr, nullptr,
           Nn, 3 * HID, HID, 0, 0.f, 0);
      gemm(H, HID, nullptr, p.gru_whh + (size_t)(l + 1) * HID * 3 * HID, 3 * HID,
           GHb, 3 * HID, nullptr, 0, p.gru_bhh + (l + 1) * 3 * HID, nullptr, nullptr,
           Nn, 3 * HID, HID, 0, 0.f, 0);
      gruk(GIb, GHb, H, Nn);
    }
    // molecule attentive readout
    fillf(GOUT, 0.f, Gg * HID);
    scatter(H, HID, nullptr, nullptr, batch, GOUT, HID, Nn);
    actk(GOUT, GOUT, Gg * HID, 2, 0.f);
    for (int it = 0; it < 3; ++it) {
      gemm(H, HID, nullptr, p.mol_w, HID, Gb, HID, nullptr, 0,
           nullptr, nullptr, nullptr, Nn, HID, HID, 0, 0.f, 0);
      gemm(GOUT, HID, nullptr, p.mol_w, HID, GD, HID, nullptr, 0,
           nullptr, nullptr, nullptr, Gg, HID, HID, 0, 0.f, 0);
      rowdot(Gb, p.mol_att_src, S1, Nn);
      rowdot(GD, p.mol_att_dst, S2, Gg);
      escore(S1, nullptr, S2, batch, Ab, Nn, 0.01f);
      segsoftmax(Ab, batch, Nn, Gg);
      bcast(GAGG, HID, p.mol_bias, Gg);
      scatter(Gb, HID, nullptr, Ab, batch, GAGG, HID, Nn);
      actk(GAGG, GX, Gg * HID, 3, 0.f);
      gemm(GX, HID, nullptr, p.mgru_wih, 3 * HID, GGI, 3 * HID, nullptr, 0,
           p.mgru_bih, nullptr, nullptr, Gg, 3 * HID, HID, 0, 0.f, 0);
      gemm(GOUT, HID, nullptr, p.mgru_whh, 3 * HID, GGH, 3 * HID, nullptr, 0,
           p.mgru_bhh, nullptr, nullptr, Gg, 3 * HID, HID, 0, 0.f, 0);
      gruk(GGI, GGH, GOUT, Gg);
    }
    gemm(GOUT, HID, nullptr, p.out_w, HID, outp, out_ld, nullptr, 0,
         p.out_b, nullptr, nullptr, Gg, HID, HID, 0, 0.f, 0);
  };

  // ================= full forward =================
  const float *sc, *sh;
  fillf(FEAT, 0.f, NMOLS * 384);   // frag_res region must start at zero

  // ---- origin channel ----
  bnprep(Pf(123), Pf(125), Pf(124), HID, &sc, &sh);  // origin.node b,g,beta
  gemm(atom_x, 44, nullptr, Pf(126), HID, NX, HID, nullptr, 0,
       nullptr, sc, sh, NATOM, HID, 44, 1, 0.01f, 0);
  bnprep(Pf(92), Pf(94), Pf(93), HID, &sc, &sh);     // origin.edge
  gemm(atom_ea, 12, nullptr, Pf(95), HID, NEA, HID, nullptr, 0,
       nullptr, sc, sh, EATOM, HID, 12, 1, 0.01f, 0);
  for (int h = 0; h < 4; ++h) {
    AfpP p = make_head(96, h, HID, HID);
    afp_run(p, NX, HID, asrc, adst, NEA, HID, abatch, NATOM, EATOM, NMOLS,
            CAT + h * HID, 4 * HID);
  }
  bnprep(Pf(88), Pf(90), Pf(89), HID, &sc, &sh);     // origin.att
  gemm(CAT, 4 * HID, nullptr, Pf(91), HID, FEAT, 384, nullptr, 0,
       nullptr, sc, sh, NMOLS, HID, 4 * HID, 2, 0.f, 0);   // -> feat[:,0:128]

  // ---- fragment channel (raw features) ----
  for (int h = 0; h < 4; ++h) {
    AfpP p = make_head(16, h, 44, 12);
    afp_run(p, frag_x, 44, fsrc, fdst, frag_ea, 12, fbatch, NFRAG, EFRAG, NFG,
            CAT + h * HID, 4 * HID);
  }
  bnprep(Pf(12), Pf(14), Pf(13), HID, &sc, &sh);     // frag.att
  gemm(CAT, 4 * HID, nullptr, Pf(15), HID, GFRAG, HID, nullptr, 0,
       nullptr, sc, sh, NFG, HID, 4 * HID, 2, 0.f, 0);

  // ---- junction channel ----
  bnprep(Pf(82), Pf(84), Pf(83), HID, &sc, &sh);     // junc.node
  gemm(junc_x, 219, nullptr, Pf(85), HID, JX + HID, 2 * HID, nullptr, 0,
       nullptr, sc, sh, NFG, HID, 219, 1, 0.01f, 0);       // motif -> jx[:,128:]
  k_copy<<<cdiv(NFG * HID, 256), 256, 0, stream>>>(GFRAG, HID, JX, 2 * HID, NFG);
  bnprep(Pf(51), Pf(53), Pf(52), HID, &sc, &sh);     // junc.edge
  gemm(junc_ea, 12, nullptr, Pf(54), HID, JEA, HID, nullptr, 0,
       nullptr, sc, sh, EJUNC, HID, 12, 1, 0.01f, 0);
  for (int h = 0; h < 4; ++h) {
    gemm(JX, 2 * HID, nullptr, Pf(87) + (size_t)h * 2 * HID * HID, HID, JH, HID,
         nullptr, 0, Pf(86) + h * HID, nullptr, nullptr, NFG, HID, 2 * HID, 0, 0.f, 0);
    AfpP p = make_head(55, h, HID, HID);
    afp_run(p, JH, HID, jsrc, jdst, JEA, HID, jbatch, NFG, EJUNC, NMOLS,
            JLAT + (size_t)h * NMOLS * HID, HID);
  }
  k_mean_relu4<<<cdiv(NMOLS * HID, 256), 256, 0, stream>>>(
      JLAT, (size_t)NMOLS * HID, FEAT + 256, 384, NMOLS);       // feat[:,256:384]
  scatter(GFRAG, HID, nullptr, nullptr, jbatch, FEAT + HID, 384, NFG); // feat[:,128:256]

  // ---- prediction head ----
  bnprep(nullptr, Pf(47), Pf(44), 192, &sc, &sh);    // head g1, beta1 (bias separate)
  gemm(FEAT, 384, nullptr, Pf(48), 192, Dh, 192, nullptr, 0,
       Pf(43), sc, sh, NMOLS, 192, 384, 1, 0.001f, 1);     // bn(lrelu(...))
  gemm(Dh, 192, nullptr, Pf(49), 192, H2, 192, nullptr, 0,
       Pf(45), nullptr, nullptr, NMOLS, 192, 192, 1, 0.001f, 1);
  gemm(H2, 192, nullptr, Pf(50), 1, (float*)d_out, 1, nullptr, 0,
       Pf(46), nullptr, nullptr, NMOLS, 1, 192, 0, 0.f, 0);
}
